// Prismatic_30571577213501
// MI455X (gfx1250) — matrix operand order repaired and checked by reading, not compiled, not run on hardware
//
#include <hip/hip_runtime.h>
#include <hip/hip_bf16.h>
#include <math.h>

#define NUM_EXPERTS 2
#define HIDDEN 768
#define BATCH 32
#define SEQ 2048

// ---- tile config ----
#define TM 128
#define TN 128
#define TK 32
#define KP 40   // padded K stride (elements) -> 80B rows: 16B aligned, conflict-free
#define KSTEPS (HIDDEN / TK)

typedef __attribute__((ext_vector_type(16))) __bf16 v16bf;
typedef __attribute__((ext_vector_type(8)))  float  v8f;
typedef __attribute__((ext_vector_type(8)))  unsigned short u16x8;
typedef __attribute__((ext_vector_type(4)))  unsigned short u16x4;
typedef int v4i __attribute__((ext_vector_type(4)));

// ---------------- CDNA5 async global->LDS (guarded) ----------------
#if __has_builtin(__builtin_amdgcn_global_load_async_to_lds_b128) && \
    __has_builtin(__builtin_amdgcn_s_wait_asynccnt)
#define HAVE_ASYNC_LDS 1
#else
#define HAVE_ASYNC_LDS 0
#endif

typedef __attribute__((address_space(1))) v4i* as1_v4i_ptr;
typedef __attribute__((address_space(3))) v4i* as3_v4i_ptr;

__device__ __forceinline__ void copy16_g2l(const unsigned short* g, unsigned short* l) {
#if HAVE_ASYNC_LDS
    __builtin_amdgcn_global_load_async_to_lds_b128(
        (as1_v4i_ptr)(void*)(g),
        (as3_v4i_ptr)(void*)(l),
        0, 0);
#else
    *(u16x8*)l = *(const u16x8*)g;
#endif
}
__device__ __forceinline__ void wait_async_lds() {
#if HAVE_ASYNC_LDS
    __builtin_amdgcn_s_wait_asynccnt(0);
#endif
}

// ---------------- helpers ----------------
__device__ __forceinline__ unsigned short f2bf_rne(float f) {
    unsigned int u = __float_as_uint(f);
    unsigned int r = u + 0x7FFFu + ((u >> 16) & 1u);   // round-to-nearest-even
    return (unsigned short)(r >> 16);
}
__device__ __forceinline__ float bf2f(unsigned short h) {
    return __uint_as_float(((unsigned int)h) << 16);
}
// jax.nn.gelu(approximate=True): 0.5x(1+tanh(u)) == x * sigmoid(2u), branch-free hw exp
__device__ __forceinline__ float gelu_tanh(float x) {
    const float c = 0.7978845608028654f; // sqrt(2/pi)
    float u = c * (x + 0.044715f * x * x * x);
    return x / (1.0f + __expf(-2.0f * u));
}

// =====================================================================
// Router: mean over seq -> LayerNorm -> linear(768x2) -> softmax/argmax
// =====================================================================
__global__ __launch_bounds__(256)
void moe_router_kernel(const float* __restrict__ x,
                       const float* __restrict__ gamma,
                       const float* __restrict__ beta,
                       const float* __restrict__ rw,   // [HIDDEN, 2]
                       const float* __restrict__ rb,   // [2]
                       int*   __restrict__ eidx,       // [BATCH]
                       float* __restrict__ probs) {    // [BATCH, 2]
    const int b = blockIdx.x;
    const int t = threadIdx.x;
    __shared__ float red[256];

    const float* xb = x + (size_t)b * SEQ * HIDDEN;
    float a0 = 0.f, a1 = 0.f, a2 = 0.f;
    for (int s = 0; s < SEQ; ++s) {
        const float* row = xb + (size_t)s * HIDDEN;
        if (s + 8 < SEQ) __builtin_prefetch(row + 8 * HIDDEN + t, 0, 1);
        a0 += row[t];
        a1 += row[t + 256];
        a2 += row[t + 512];
    }
    const float inv_seq = 1.0f / (float)SEQ;
    float m0 = a0 * inv_seq, m1 = a1 * inv_seq, m2 = a2 * inv_seq;

    // mean over hidden
    red[t] = m0 + m1 + m2;
    __syncthreads();
    for (int off = 128; off > 0; off >>= 1) {
        if (t < off) red[t] += red[t + off];
        __syncthreads();
    }
    const float mu = red[0] * (1.0f / (float)HIDDEN);
    __syncthreads();

    // biased var (jnp.var)
    red[t] = m0*m0 + m1*m1 + m2*m2;
    __syncthreads();
    for (int off = 128; off > 0; off >>= 1) {
        if (t < off) red[t] += red[t + off];
        __syncthreads();
    }
    const float var = red[0] * (1.0f / (float)HIDDEN) - mu * mu;
    __syncthreads();

    const float rstd = rsqrtf(var + 1e-5f);
    float n0 = (m0 - mu) * rstd * gamma[t]       + beta[t];
    float n1 = (m1 - mu) * rstd * gamma[t + 256] + beta[t + 256];
    float n2 = (m2 - mu) * rstd * gamma[t + 512] + beta[t + 512];

    float l0 = n0 * rw[t*2 + 0] + n1 * rw[(t+256)*2 + 0] + n2 * rw[(t+512)*2 + 0];
    float l1 = n0 * rw[t*2 + 1] + n1 * rw[(t+256)*2 + 1] + n2 * rw[(t+512)*2 + 1];

    red[t] = l0;
    __syncthreads();
    for (int off = 128; off > 0; off >>= 1) {
        if (t < off) red[t] += red[t + off];
        __syncthreads();
    }
    const float L0r = red[0];
    __syncthreads();
    red[t] = l1;
    __syncthreads();
    for (int off = 128; off > 0; off >>= 1) {
        if (t < off) red[t] += red[t + off];
        __syncthreads();
    }
    const float L1r = red[0];

    if (t == 0) {
        float L0 = L0r + rb[0];
        float L1 = L1r + rb[1];
        float mx = fmaxf(L0, L1);
        float e0 = expf(L0 - mx), e1 = expf(L1 - mx);
        float z = e0 + e1;
        probs[2*b + 0] = e0 / z;
        probs[2*b + 1] = e1 / z;
        eidx[b] = (L0 >= L1) ? 0 : 1;   // jnp.argmax tie -> first
    }
}

// =====================================================================
// Aux loss: 0.01 * mean_e((mean_b probs - 0.5)^2)
// =====================================================================
__global__ void moe_aux_kernel(const float* __restrict__ probs,
                               float* __restrict__ out_loss) {
    float a0 = 0.f, a1 = 0.f;
    for (int b = 0; b < BATCH; ++b) { a0 += probs[2*b]; a1 += probs[2*b + 1]; }
    a0 *= (1.0f / BATCH); a1 *= (1.0f / BATCH);
    float d0 = a0 - 0.5f, d1 = a1 - 0.5f;
    out_loss[0] = 0.01f * 0.5f * (d0*d0 + d1*d1);
}

// =====================================================================
// Pre-conversion: X -> bf16 hi/lo (same layout)
// =====================================================================
__global__ __launch_bounds__(256)
void convert_x_kernel(const float* __restrict__ x,
                      unsigned short* __restrict__ xh,
                      unsigned short* __restrict__ xl) {
    const size_t n4 = (size_t)BATCH * SEQ * HIDDEN / 4;
    const size_t stride = (size_t)gridDim.x * blockDim.x;
    for (size_t i = (size_t)blockIdx.x * blockDim.x + threadIdx.x; i < n4; i += stride) {
        float4 v = ((const float4*)x)[i];
        float fs[4] = {v.x, v.y, v.z, v.w};
        u16x4 h, l;
        #pragma unroll
        for (int j = 0; j < 4; ++j) {
            unsigned short hb = f2bf_rne(fs[j]);
            h[j] = hb;
            l[j] = f2bf_rne(fs[j] - bf2f(hb));
        }
        ((u16x4*)xh)[i] = h;
        ((u16x4*)xl)[i] = l;
    }
}

// =====================================================================
// Pre-conversion: W[e][k][n] -> transposed bf16 hi/lo Wt[e][n][k]
// =====================================================================
__global__ __launch_bounds__(256)
void convert_w_kernel(const float* __restrict__ w,
                      unsigned short* __restrict__ wth,
                      unsigned short* __restrict__ wtl) {
    const size_t total4 = (size_t)NUM_EXPERTS * HIDDEN * HIDDEN / 4;
    const size_t stride = (size_t)gridDim.x * blockDim.x;
    for (size_t i = (size_t)blockIdx.x * blockDim.x + threadIdx.x; i < total4; i += stride) {
        int k4  = (int)(i % (HIDDEN / 4)) * 4;
        int tmp = (int)(i / (HIDDEN / 4));
        int n   = tmp % HIDDEN;
        int e   = tmp / HIDDEN;
        const size_t base = (size_t)e * HIDDEN * HIDDEN;
        u16x4 h, l;
        #pragma unroll
        for (int j = 0; j < 4; ++j) {
            float f = w[base + (size_t)(k4 + j) * HIDDEN + n];
            unsigned short hb = f2bf_rne(f);
            h[j] = hb;
            l[j] = f2bf_rne(f - bf2f(hb));
        }
        *(u16x4*)(wth + base + (size_t)n * HIDDEN + k4) = h;
        *(u16x4*)(wtl + base + (size_t)n * HIDDEN + k4) = l;
    }
}

union FragU { v16bf v; u16x8 h[2]; };

// =====================================================================
// GEMM from pre-converted bf16 hi/lo, async global->LDS, double-buffered
// Block: 256 thr (8 wave32), 128x128 tile; wave: 32x64 (2x4 16x16 tiles)
// =====================================================================
__global__ __launch_bounds__(256)
void moe_expert_gemm_pre(const unsigned short* __restrict__ Xh,
                         const unsigned short* __restrict__ Xl,
                         const unsigned short* __restrict__ Wth,  // [E][n][k]
                         const unsigned short* __restrict__ Wtl,
                         const float* __restrict__ ebias,
                         const int*   __restrict__ eidx,
                         float* __restrict__ out) {
    // [buf][0=Ah,1=Al,2=Bh,3=Bl][128*KP]
    __shared__ __attribute__((aligned(16))) unsigned short smem[2][4][TM * KP];

    const int t    = threadIdx.x;
    const int mblk = blockIdx.x * TM;
    const int nblk = blockIdx.y * TN;
    const int b    = blockIdx.z;
    const int e    = eidx[b];

    const unsigned short* srcAh = Xh + ((size_t)b * SEQ + mblk) * HIDDEN;
    const unsigned short* srcAl = Xl + ((size_t)b * SEQ + mblk) * HIDDEN;
    const unsigned short* srcBh = Wth + (size_t)e * HIDDEN * HIDDEN + (size_t)nblk * HIDDEN;
    const unsigned short* srcBl = Wtl + (size_t)e * HIDDEN * HIDDEN + (size_t)nblk * HIDDEN;
    const float* bias = ebias + (size_t)e * HIDDEN;

    const int lane    = t & 31;
    const int wave    = t >> 5;
    const int m_wave  = (wave & 3) * 32;   // 4 waves along M
    const int n_wave  = (wave >> 2) * 64;  // 2 waves along N
    const int lm      = lane & 15;
    const int hi_half = lane >> 4;

    v8f acc[2][4];
    #pragma unroll
    for (int i = 0; i < 2; ++i)
        #pragma unroll
        for (int j = 0; j < 4; ++j) {
            v8f z = {};
            acc[i][j] = z;
        }

    // stage one K-step tile (A hi/lo 128x32, B hi/lo 128x32) into LDS buffer
    auto stage = [&](int buf, int k0) {
        #pragma unroll
        for (int i = 0; i < 2; ++i) {
            int id  = t + i * 256;       // 0..511
            int row = id >> 2;           // 0..127 (A row / B column)
            int seg = (id & 3) * 8;      // element offset: 0,8,16,24 (16B chunks)
            size_t g = (size_t)row * HIDDEN + k0 + seg;
            int    l = row * KP + seg;
            copy16_g2l(srcAh + g, &smem[buf][0][l]);
            copy16_g2l(srcAl + g, &smem[buf][1][l]);
            copy16_g2l(srcBh + g, &smem[buf][2][l]);
            copy16_g2l(srcBl + g, &smem[buf][3][l]);
        }
    };

    stage(0, 0);
    for (int kt = 0; kt < KSTEPS; ++kt) {
        wait_async_lds();       // my async writes to buf (kt&1) done
        __syncthreads();        // everyone's writes visible; prev compute done
        if (kt + 1 < KSTEPS) stage((kt + 1) & 1, (kt + 1) * TK);
        const int buf = kt & 1;

        // ---- fragments per ISA 7.12.2 lane layouts ----
        FragU ah[2], al[2], bh[4], bl[4];
        #pragma unroll
        for (int mt = 0; mt < 2; ++mt) {
            int row   = m_wave + mt * 16 + lm;
            int kbase = hi_half * 8;
            ah[mt].h[0] = *(const u16x8*)&smem[buf][0][row * KP + kbase];
            ah[mt].h[1] = *(const u16x8*)&smem[buf][0][row * KP + kbase + 16];
            al[mt].h[0] = *(const u16x8*)&smem[buf][1][row * KP + kbase];
            al[mt].h[1] = *(const u16x8*)&smem[buf][1][row * KP + kbase + 16];
        }
        #pragma unroll
        for (int nt = 0; nt < 4; ++nt) {
            int col   = n_wave + nt * 16 + lm;
            int kbase = hi_half * 8;
            bh[nt].h[0] = *(const u16x8*)&smem[buf][2][col * KP + kbase];
            bh[nt].h[1] = *(const u16x8*)&smem[buf][2][col * KP + kbase + 16];
            bl[nt].h[0] = *(const u16x8*)&smem[buf][3][col * KP + kbase];
            bl[nt].h[1] = *(const u16x8*)&smem[buf][3][col * KP + kbase + 16];
        }

        // ---- bf16x3: Ah*Bh + Ah*Bl + Al*Bh, f32 accumulate ----
        #pragma unroll
        for (int mt = 0; mt < 2; ++mt) {
            #pragma unroll
            for (int nt = 0; nt < 4; ++nt) {
                acc[mt][nt] = __builtin_amdgcn_wmma_f32_16x16x32_bf16(
                    false, ah[mt].v, false, bh[nt].v, (short)0, acc[mt][nt], false, false);
                acc[mt][nt] = __builtin_amdgcn_wmma_f32_16x16x32_bf16(
                    false, ah[mt].v, false, bl[nt].v, (short)0, acc[mt][nt], false, false);
                acc[mt][nt] = __builtin_amdgcn_wmma_f32_16x16x32_bf16(
                    false, al[mt].v, false, bh[nt].v, (short)0, acc[mt][nt], false, false);
            }
        }
    }

    // ---- epilogue: bias + GELU + store ----
    float* ob = out + (size_t)b * SEQ * HIDDEN;
    #pragma unroll
    for (int nt = 0; nt < 4; ++nt) {
        int col  = nblk + n_wave + nt * 16 + lm;
        float bv = bias[col];
        #pragma unroll
        for (int mt = 0; mt < 2; ++mt) {
            int rbase = mblk + m_wave + mt * 16 + hi_half * 8;
            #pragma unroll
            for (int r = 0; r < 8; ++r) {
                float v = acc[mt][nt][r] + bv;
                ob[(size_t)(rbase + r) * HIDDEN + col] = gelu_tanh(v);
            }
        }
    }
}

// =====================================================================
// Fallback fused GEMM (fp32 in global, convert in-kernel) — small ws
// =====================================================================
__global__ __launch_bounds__(256)
void moe_expert_gemm_fused(const float* __restrict__ x,
                           const float* __restrict__ ew,
                           const float* __restrict__ ebias,
                           const int*   __restrict__ eidx,
                           float* __restrict__ out) {
    __shared__ __attribute__((aligned(16))) unsigned short Ah[TM * KP];
    __shared__ __attribute__((aligned(16))) unsigned short Al[TM * KP];
    __shared__ __attribute__((aligned(16))) unsigned short Bh[TN * KP];
    __shared__ __attribute__((aligned(16))) unsigned short Bl[TN * KP];

    const int t    = threadIdx.x;
    const int mblk = blockIdx.x * TM;
    const int nblk = blockIdx.y * TN;
    const int b    = blockIdx.z;
    const int e    = eidx[b];

    const float* xb   = x + (size_t)b * SEQ * HIDDEN;
    const float* W    = ew + (size_t)e * HIDDEN * HIDDEN;
    const float* bias = ebias + (size_t)e * HIDDEN;

    const int lane    = t & 31;
    const int wave    = t >> 5;
    const int m_wave  = (wave & 3) * 32;
    const int n_wave  = (wave >> 2) * 64;
    const int lm      = lane & 15;
    const int hi_half = lane >> 4;

    v8f acc[2][4];
    #pragma unroll
    for (int i = 0; i < 2; ++i)
        #pragma unroll
        for (int j = 0; j < 4; ++j) {
            v8f z = {};
            acc[i][j] = z;
        }

    for (int k0 = 0; k0 < HIDDEN; k0 += TK) {
        __syncthreads();
        #pragma unroll
        for (int i = 0; i < 4; ++i) {
            int id  = t + i * 256;
            int row = id >> 3;
            int c4  = id & 7;
            const float4 v = *(const float4*)(xb + (size_t)(mblk + row) * HIDDEN + k0 + c4 * 4);
            float fs[4] = {v.x, v.y, v.z, v.w};
            u16x4 h, l;
            #pragma unroll
            for (int j = 0; j < 4; ++j) {
                unsigned short hb = f2bf_rne(fs[j]);
                h[j] = hb;
                l[j] = f2bf_rne(fs[j] - bf2f(hb));
            }
            *(u16x4*)&Ah[row * KP + c4 * 4] = h;
            *(u16x4*)&Al[row * KP + c4 * 4] = l;
        }
        #pragma unroll
        for (int i = 0; i < 4; ++i) {
            int id = t + i * 256;
            int kk = id >> 5;
            int c4 = id & 31;
            const float4 v = *(const float4*)(W + (size_t)(k0 + kk) * HIDDEN + nblk + c4 * 4);
            float fs[4] = {v.x, v.y, v.z, v.w};
            #pragma unroll
            for (int j = 0; j < 4; ++j) {
                unsigned short hb = f2bf_rne(fs[j]);
                Bh[(c4 * 4 + j) * KP + kk] = hb;
                Bl[(c4 * 4 + j) * KP + kk] = f2bf_rne(fs[j] - bf2f(hb));
            }
        }
        __syncthreads();

        FragU ah[2], al[2], bh[4], bl[4];
        #pragma unroll
        for (int mt = 0; mt < 2; ++mt) {
            int row   = m_wave + mt * 16 + lm;
            int kbase = hi_half * 8;
            ah[mt].h[0] = *(const u16x8*)&Ah[row * KP + kbase];
            ah[mt].h[1] = *(const u16x8*)&Ah[row * KP + kbase + 16];
            al[mt].h[0] = *(const u16x8*)&Al[row * KP + kbase];
            al[mt].h[1] = *(const u16x8*)&Al[row * KP + kbase + 16];
        }
        #pragma unroll
        for (int nt = 0; nt < 4; ++nt) {
            int col   = n_wave + nt * 16 + lm;
            int kbase = hi_half * 8;
            bh[nt].h[0] = *(const u16x8*)&Bh[col * KP + kbase];
            bh[nt].h[1] = *(const u16x8*)&Bh[col * KP + kbase + 16];
            bl[nt].h[0] = *(const u16x8*)&Bl[col * KP + kbase];
            bl[nt].h[1] = *(const u16x8*)&Bl[col * KP + kbase + 16];
        }
        #pragma unroll
        for (int mt = 0; mt < 2; ++mt) {
            #pragma unroll
            for (int nt = 0; nt < 4; ++nt) {
                acc[mt][nt] = __builtin_amdgcn_wmma_f32_16x16x32_bf16(
                    false, ah[mt].v, false, bh[nt].v, (short)0, acc[mt][nt], false, false);
                acc[mt][nt] = __builtin_amdgcn_wmma_f32_16x16x32_bf16(
                    false, ah[mt].v, false, bl[nt].v, (short)0, acc[mt][nt], false, false);
                acc[mt][nt] = __builtin_amdgcn_wmma_f32_16x16x32_bf16(
                    false, al[mt].v, false, bh[nt].v, (short)0, acc[mt][nt], false, false);
            }
        }
    }

    float* ob = out + (size_t)b * SEQ * HIDDEN;
    #pragma unroll
    for (int nt = 0; nt < 4; ++nt) {
        int col  = nblk + n_wave + nt * 16 + lm;
        float bv = bias[col];
        #pragma unroll
        for (int mt = 0; mt < 2; ++mt) {
            int rbase = mblk + m_wave + mt * 16 + hi_half * 8;
            #pragma unroll
            for (int r = 0; r < 8; ++r) {
                float v = acc[mt][nt][r] + bv;
                ob[(size_t)(rbase + r) * HIDDEN + col] = gelu_tanh(v);
            }
        }
    }
}

// =====================================================================
extern "C" void kernel_launch(void* const* d_in, const int* in_sizes, int n_in,
                              void* d_out, int out_size, void* d_ws, size_t ws_size,
                              hipStream_t stream) {
    const float* x     = (const float*)d_in[0];
    const float* gamma = (const float*)d_in[1];
    const float* beta  = (const float*)d_in[2];
    const float* rw    = (const float*)d_in[3];
    const float* rb    = (const float*)d_in[4];
    const float* ew    = (const float*)d_in[5];
    const float* eb    = (const float*)d_in[6];
    float* out = (float*)d_out;

    int*   eidx  = (int*)d_ws;
    float* probs = (float*)((char*)d_ws + 128);

    moe_router_kernel<<<BATCH, 256, 0, stream>>>(x, gamma, beta, rw, rb, eidx, probs);
    moe_aux_kernel<<<1, 1, 0, stream>>>(probs, out + (size_t)BATCH * SEQ * HIDDEN);

    const size_t xsz = (size_t)BATCH * SEQ * HIDDEN;          // elements
    const size_t wsz = (size_t)NUM_EXPERTS * HIDDEN * HIDDEN; // elements
    const size_t need = 1024 + 2 * xsz * 2 + 2 * wsz * 2;     // bytes (hi+lo bf16)

    dim3 grid(SEQ / TM, HIDDEN / TN, BATCH);

    if (ws_size >= need) {
        unsigned short* Xh  = (unsigned short*)((char*)d_ws + 1024);
        unsigned short* Xl  = Xh + xsz;
        unsigned short* Wth = Xl + xsz;
        unsigned short* Wtl = Wth + wsz;
        convert_x_kernel<<<4096, 256, 0, stream>>>(x, Xh, Xl);
        convert_w_kernel<<<1152, 256, 0, stream>>>(ew, Wth, Wtl);
        moe_expert_gemm_pre<<<grid, 256, 0, stream>>>(Xh, Xl, Wth, Wtl, eb, eidx, out);
    } else {
        moe_expert_gemm_fused<<<grid, 256, 0, stream>>>(x, ew, eb, eidx, out);
    }
}
